// GraphAttentionLayer_72541997629931
// MI455X (gfx1250) — matrix operand order repaired and checked by reading, not compiled, not run on hardware
//
#include <hip/hip_runtime.h>
#include <hip/hip_bf16.h>

// Problem constants (match reference)
#define BB    8
#define NN    2048
#define DIN_  256
#define DOUT_ 256
#define ALPHA_ 0.2f
#define EPS_   1e-6f

typedef __attribute__((ext_vector_type(16))) __bf16 v16bf;
typedef __attribute__((ext_vector_type(8)))  float  v8f;
typedef __attribute__((ext_vector_type(4)))  int    v4i_;

union AFrag16 { v16bf v; unsigned short u[16]; };
union BFrag16 { v16bf v; uint4 q[2]; };

// CDNA5 async global->LDS copy path (guarded; falls back to vector copies)
#if __has_builtin(__builtin_amdgcn_global_load_async_to_lds_b128) && \
    __has_builtin(__builtin_amdgcn_s_wait_asynccnt)
#define USE_ASYNC_LDS 1
typedef __attribute__((address_space(1))) v4i_ gv4i;   // global int4 vector
typedef __attribute__((address_space(3))) v4i_ lv4i;   // LDS int4 vector
#else
#define USE_ASYNC_LDS 0
#endif

// float -> bf16 bits via native conversion (RNE on hardware with cvt)
__device__ __forceinline__ unsigned short f2bf(float f) {
  union { __bf16 b; unsigned short u; } c;
  c.b = (__bf16)f;
  return c.u;
}

// ---------------------------------------------------------------------------
// Kernel 1: Wh = h @ W (bf16 WMMA, f32 acc); emits Wh TRANSPOSED as bf16
// (WhT_g[b][f][j]) for kernel 2's B-operand, plus s1 = Wh@a1, s2 = Wh@a2.
// Grid: B * (N/128) blocks of 256 threads (8 waves); wave owns 16 rows.
// ---------------------------------------------------------------------------
__global__ __launch_bounds__(256) void gat_wh_kernel(
    const float* __restrict__ h, const float* __restrict__ W,
    const float* __restrict__ a,
    __bf16* __restrict__ WhTg, float* __restrict__ s1, float* __restrict__ s2)
{
  // W^T bf16 [f][k], row stride 264 u16 = 528B (16B multiple)
  __shared__ __align__(16) unsigned short WT[DOUT_ * 264];

  const int tid  = threadIdx.x;
  const int wave = tid >> 5;
  const int lane = tid & 31;
  const int hi   = lane >> 4;      // half-wave selector
  const int row  = lane & 15;      // A lane row / B lane col index
  const int b    = blockIdx.x / (NN / 128);
  const int i0   = (blockIdx.x % (NN / 128)) * 128 + wave * 16;

  // Stage W^T as bf16: thread t owns output column f = t (coalesced reads).
  {
    const int f = tid;
    for (int k = 0; k < DIN_; ++k)
      WT[f * 264 + k] = f2bf(W[k * DOUT_ + f]);
  }
  __syncthreads();

  // All 8 A fragments (16x32 bf16) for this wave's 16 rows.
  // A layout: lane holds row (lane&15); elems 0..7 -> K = k0+hi*8+e,
  //           elems 8..15 -> K = k0+16+hi*8+(e-8).
  AFrag16 afr[8];
  const float* hrow = h + ((size_t)b * NN + (i0 + row)) * DIN_;
#pragma unroll
  for (int ks = 0; ks < 8; ++ks) {
    const int k0 = ks * 32;
    float4 r0 = *(const float4*)(hrow + k0 + hi * 8);
    float4 r1 = *(const float4*)(hrow + k0 + hi * 8 + 4);
    float4 r2 = *(const float4*)(hrow + k0 + 16 + hi * 8);
    float4 r3 = *(const float4*)(hrow + k0 + 16 + hi * 8 + 4);
    afr[ks].u[0]  = f2bf(r0.x); afr[ks].u[1]  = f2bf(r0.y);
    afr[ks].u[2]  = f2bf(r0.z); afr[ks].u[3]  = f2bf(r0.w);
    afr[ks].u[4]  = f2bf(r1.x); afr[ks].u[5]  = f2bf(r1.y);
    afr[ks].u[6]  = f2bf(r1.z); afr[ks].u[7]  = f2bf(r1.w);
    afr[ks].u[8]  = f2bf(r2.x); afr[ks].u[9]  = f2bf(r2.y);
    afr[ks].u[10] = f2bf(r2.z); afr[ks].u[11] = f2bf(r2.w);
    afr[ks].u[12] = f2bf(r3.x); afr[ks].u[13] = f2bf(r3.y);
    afr[ks].u[14] = f2bf(r3.z); afr[ks].u[15] = f2bf(r3.w);
  }

  const float* a1 = a;
  const float* a2 = a + DOUT_;
  float s1p[8] = {0.f,0.f,0.f,0.f,0.f,0.f,0.f,0.f};
  float s2p[8] = {0.f,0.f,0.f,0.f,0.f,0.f,0.f,0.f};
  __bf16* wtg = WhTg + (size_t)b * DOUT_ * NN;

#pragma unroll 1
  for (int ct = 0; ct < 16; ++ct) {
    v8f acc = {};
    const int col = ct * 16 + row;                 // B column == C column
    const unsigned short* wcol = &WT[col * 264];
#pragma unroll
    for (int ks = 0; ks < 8; ++ks) {
      BFrag16 bfr;
      // B layout: lane holds col; elems 0..7 -> K = ks*32 + hi*8 + e, 8..15 -> K = ks*32 + 16 + hi*8 + (e-8)
      bfr.q[0] = *(const uint4*)(wcol + ks * 32 + hi * 8);
      bfr.q[1] = *(const uint4*)(wcol + ks * 32 + 16 + hi * 8);
      acc = __builtin_amdgcn_wmma_f32_16x16x32_bf16(
          false, afr[ks].v, false, bfr.v, (short)0, acc, false, false);
    }
    const float a1c = a1[col], a2c = a2[col];
    // C layout: VGPR r, lane -> row r+8*hi. Rows are contiguous j in WhT_g:
    // pack 8 bf16 and emit one b128 store per lane.
    union { unsigned short us[8]; uint4 q; } pk;
#pragma unroll
    for (int r = 0; r < 8; ++r) {
      const float v = acc[r];
      pk.us[r] = f2bf(v);
      s1p[r] += v * a1c;
      s2p[r] += v * a2c;
    }
    *(uint4*)(wtg + (size_t)col * NN + i0 + 8 * hi) = pk.q;
  }

  // Reduce s-partials across each 16-lane half (cols); rows are r+8*hi.
#pragma unroll
  for (int m = 1; m < 16; m <<= 1) {
#pragma unroll
    for (int r = 0; r < 8; ++r) {
      s1p[r] += __shfl_xor(s1p[r], m, 32);
      s2p[r] += __shfl_xor(s2p[r], m, 32);
    }
  }
  if (row == 0) {
#pragma unroll
    for (int r = 0; r < 8; ++r) {
      s1[(size_t)b * NN + i0 + 8 * hi + r] = s1p[r];
      s2[(size_t)b * NN + i0 + 8 * hi + r] = s2p[r];
    }
  }
}

// ---------------------------------------------------------------------------
// Kernel 2: fused flash-style attention. Per wave: 16-row i-tile, online
// softmax over j in 32-chunks, P@Wh via bf16 WMMA into 16x256 f32 acc.
// Wh^T bf16 chunk staged into double-buffered LDS (async copy when available).
// ---------------------------------------------------------------------------
__global__ __launch_bounds__(256) void gat_attn_kernel(
    const float* __restrict__ adj, const __bf16* __restrict__ WhTg,
    const float* __restrict__ s1g, const float* __restrict__ s2g,
    float* __restrict__ out)
{
  // [buf][f][j] bf16; row stride 40 u16 = 80B (16B multiple)
  __shared__ __align__(16) unsigned short WhT[2][DOUT_ * 40];
  __shared__ float s2s[NN];

  const int tid  = threadIdx.x;
  const int wave = tid >> 5;
  const int lane = tid & 31;
  const int hi   = lane >> 4;
  const int row  = lane & 15;
  const int b    = blockIdx.x / (NN / 128);
  const int i0   = (blockIdx.x % (NN / 128)) * 128 + wave * 16;

  for (int j = tid; j < NN; j += 256) s2s[j] = s2g[(size_t)b * NN + j];

  const float s1r = s1g[(size_t)b * NN + i0 + row];
  const float* adjrow = adj + ((size_t)b * NN + (i0 + row)) * NN;
  const __bf16* whtb = WhTg + (size_t)b * DOUT_ * NN;

  // Stage one 32-wide j chunk: thread t copies row f=t (4 x b128).
  auto stage = [&](int bufsel, int j0s) {
    const uint4* src = (const uint4*)(whtb + (size_t)tid * NN + j0s);
    uint4* dst = (uint4*)&WhT[bufsel][tid * 40];
#if USE_ASYNC_LDS
#pragma unroll
    for (int c = 0; c < 4; ++c)
      __builtin_amdgcn_global_load_async_to_lds_b128(
          (gv4i*)(uintptr_t)(src + c),
          (lv4i*)(unsigned)(uintptr_t)(dst + c), 0, 0);
#else
#pragma unroll
    for (int c = 0; c < 4; ++c) dst[c] = src[c];
#endif
  };

  float mrun = -3.0e38f, lrun = 0.f;
  v8f acc[16] = {};

  stage(0, 0);
#if USE_ASYNC_LDS
  __builtin_amdgcn_s_wait_asynccnt(0);
#endif
  __syncthreads();

#pragma unroll 1
  for (int jc = 0; jc < NN / 32; ++jc) {
    const int j0 = jc * 32;
    const unsigned short* cur = WhT[jc & 1];

    // Kick off next chunk's copy into the other buffer (overlaps compute).
    if (jc + 1 < NN / 32) {
      stage((jc + 1) & 1, j0 + 32);
      __builtin_prefetch(adjrow + j0 + 32, 0, 3);  // global_prefetch_b8 for adj
    }

    // adj tile in A-fragment order (two 8-wide runs per half-wave).
    float4 q0 = *(const float4*)(adjrow + j0 + hi * 8);
    float4 q1 = *(const float4*)(adjrow + j0 + hi * 8 + 4);
    float4 q2 = *(const float4*)(adjrow + j0 + 16 + hi * 8);
    float4 q3 = *(const float4*)(adjrow + j0 + 16 + hi * 8 + 4);
    float av[16] = {q0.x,q0.y,q0.z,q0.w, q1.x,q1.y,q1.z,q1.w,
                    q2.x,q2.y,q2.z,q2.w, q3.x,q3.y,q3.z,q3.w};

    float e[16];
    float tmax = -3.0e38f;
#pragma unroll
    for (int t = 0; t < 16; ++t) {
      const int K = (t < 8) ? (hi * 8 + t) : (16 + hi * 8 + (t - 8));
      float ev = s1r + s2s[j0 + K];
      ev = fmaxf(ev, ALPHA_ * ev);              // LeakyReLU
      ev += __logf(av[t] + EPS_);
      e[t] = ev;
      tmax = fmaxf(tmax, ev);
    }
    tmax = fmaxf(tmax, __shfl_xor(tmax, 16, 32));  // combine row halves

    const float mnew  = fmaxf(mrun, tmax);
    const float scale = __expf(mrun - mnew);       // 0 on first chunk
    float tsum = 0.f;
    AFrag16 p;
#pragma unroll
    for (int t = 0; t < 16; ++t) {
      const float pv = __expf(e[t] - mnew);
      tsum += pv;
      p.u[t] = f2bf(pv);
    }
    tsum += __shfl_xor(tsum, 16, 32);
    lrun = lrun * scale + tsum;
    mrun = mnew;

    // Rescale accumulators only when some row's max moved (scale != 1).
    if (__any(scale < 1.f)) {
      float scv[8];
#pragma unroll
      for (int r = 0; r < 8; ++r) scv[r] = __shfl(scale, r + 8 * hi, 32);
#pragma unroll
      for (int ct = 0; ct < 16; ++ct)
#pragma unroll
        for (int r = 0; r < 8; ++r) acc[ct][r] *= scv[r];
    }

    // 16 col tiles of P(16x32) @ Wh(32x16)
#pragma unroll
    for (int ct = 0; ct < 16; ++ct) {
      BFrag16 bfr;
      const unsigned short* base = &cur[(ct * 16 + row) * 40 + hi * 8];
      bfr.q[0] = *(const uint4*)(base);
      bfr.q[1] = *(const uint4*)(base + 16);
      acc[ct] = __builtin_amdgcn_wmma_f32_16x16x32_bf16(
          false, p.v, false, bfr.v, (short)0, acc[ct], false, false);
    }

#if USE_ASYNC_LDS
    __builtin_amdgcn_s_wait_asynccnt(0);
#endif
    __syncthreads();   // next-chunk staging complete; safe to swap buffers
  }

  // Normalize and write out (coalesced 16-lane rows per (r, ct)).
  float lr[8];
#pragma unroll
  for (int r = 0; r < 8; ++r) lr[r] = __shfl(lrun, r + 8 * hi, 32);
  float* orow = out + ((size_t)b * NN + i0) * DOUT_;
#pragma unroll
  for (int ct = 0; ct < 16; ++ct)
#pragma unroll
    for (int r = 0; r < 8; ++r)
      orow[(size_t)(r + 8 * hi) * DOUT_ + ct * 16 + row] = acc[ct][r] / lr[r];
}

// ---------------------------------------------------------------------------
extern "C" void kernel_launch(void* const* d_in, const int* in_sizes, int n_in,
                              void* d_out, int out_size, void* d_ws, size_t ws_size,
                              hipStream_t stream) {
  const float* h   = (const float*)d_in[0];   // (8,2048,256)
  const float* adj = (const float*)d_in[1];   // (8,2048,2048)
  const float* W   = (const float*)d_in[2];   // (256,256)
  const float* a   = (const float*)d_in[3];   // (512,1)
  float* out = (float*)d_out;                 // (8,2048,256)

  __bf16* WhTg = (__bf16*)d_ws;               // 8*256*2048 bf16 = 8 MB
  float* s1 = (float*)((char*)d_ws + (size_t)BB * DOUT_ * NN * sizeof(__bf16));
  float* s2 = s1 + (size_t)BB * NN;

  const int blocks = BB * (NN / 128);         // 128
  gat_wh_kernel<<<blocks, 256, 0, stream>>>(h, W, a, WhTg, s1, s2);
  gat_attn_kernel<<<blocks, 256, 0, stream>>>(adj, WhTg, s1, s2, out);
}
